// GNNEncoder_20633022890119
// MI455X (gfx1250) — matrix operand order repaired and checked by reading, not compiled, not run on hardware
//
#include <hip/hip_runtime.h>

// ---------------------------------------------------------------------------
// GNN encoder (GINEConv x2 + BN + mean-pool + proj) for gfx1250 (MI455X).
//  - Edge embeddings recomputed per 64-edge block (no [E,H] materialization):
//    one v_wmma_f32_16x16x32_bf16 per 16-edge tile, gather h[src], relu,
//    scatter with f32 atomics into L2-resident agg.
//  - Dense GEMMs: bf16 WMMA, f32 accum. 64-row A tiles staged via LDS with
//    inline f32->bf16 convert (+ fused h+agg add); B pre-transposed bf16.
//    B-fragment reused across 4 row-tiles (4 independent accumulators).
//  - global_prefetch_b8 issued for gather rows while the WMMA executes.
// ---------------------------------------------------------------------------

typedef __attribute__((ext_vector_type(16))) __bf16 bf16x16;
typedef __attribute__((ext_vector_type(8)))  float  f32x8;

#define HDIM 128

// Convert weight [K][H] f32 (row-major) -> transposed bf16 [H][Kpad], zero pad.
__global__ void cvt_w_kernel(const float* __restrict__ src, __bf16* __restrict__ dst,
                             int K, int Kpad) {
    int idx = blockIdx.x * blockDim.x + threadIdx.x;
    int total = HDIM * Kpad;
    if (idx >= total) return;
    int col = idx / Kpad;
    int k   = idx - col * Kpad;
    float v = (k < K) ? src[(size_t)k * HDIM + col] : 0.0f;
    dst[idx] = (__bf16)v;
}

// Generic WMMA GEMM: out[M,H] = act( (A0 (+A1)) @ Bt^T + bias ), H=128.
// A0/A1: f32 [M,K]; Bt: bf16 [H][K] (transposed weights); K multiple of 32.
// Block = 8 waves, 64 rows (4 row-tiles); wave w owns cols [16w,16w+16).
__global__ __launch_bounds__(256)
void gemm_wmma_kernel(const float* __restrict__ A0, const float* __restrict__ A1,
                      const __bf16* __restrict__ Bt, const float* __restrict__ bias,
                      float* __restrict__ out, int M, int K, int do_relu) {
    __shared__ __bf16 ldsA[64 * 128];   // 64 rows x K (<=128) bf16 = 16 KB
    const int tid  = threadIdx.x;
    const int rowB = blockIdx.x * 64;

    // Stage A tile into LDS with f32->bf16 convert (and optional A0+A1 fuse).
    for (int idx = tid; idx < 64 * K; idx += 256) {
        int r = idx / K;
        int k = idx - r * K;
        int row = rowB + r;
        float v = 0.0f;
        if (row < M) {
            size_t off = (size_t)row * K + k;
            v = A0[off];
            if (A1) v += A1[off];
        }
        ldsA[idx] = (__bf16)v;
    }
    __syncthreads();

    const int wave = tid >> 5, lane = tid & 31;
    const int half = lane >> 4, l16 = lane & 15;
    const int col  = wave * 16 + l16;

    f32x8 acc[4] = {};
    for (int kk = 0; kk < K; kk += 32) {
        bf16x16 b;
#pragma unroll
        for (int i = 0; i < 16; ++i)                           // contiguous k = 16*half + i (same as A)
            b[i] = Bt[(size_t)col * K + kk + 16 * half + i];
#pragma unroll
        for (int mt = 0; mt < 4; ++mt) {
            bf16x16 a;
#pragma unroll
            for (int i = 0; i < 16; ++i) {
                int ka = kk + 16 * half + i;                    // contiguous k = 16*half + i (same as B)
                a[i] = ldsA[(mt * 16 + l16) * K + ka];
            }
            acc[mt] = __builtin_amdgcn_wmma_f32_16x16x32_bf16(false, a, false, b,
                                                              (short)0, acc[mt],
                                                              false, false);
        }
    }

    const float bv = bias ? bias[col] : 0.0f;
#pragma unroll
    for (int mt = 0; mt < 4; ++mt) {
#pragma unroll
        for (int r = 0; r < 8; ++r) {
            int row = rowB + mt * 16 + r + 8 * half;           // ISA C/D layout
            if (row < M) {
                float v = acc[mt][r] + bv;
                if (do_relu) v = fmaxf(v, 0.0f);
                out[(size_t)row * HDIM + col] = v;
            }
        }
    }
}

// Fused GINE message pass, 64 edges per block (4 tiles of 16):
//   e   = edge_attr(16x16, K zero-padded to 32) @ edge_w  (1 WMMA/tile/wave)
//   msg = relu(h[src] + e + edge_b) ; agg[dst] += msg (f32 atomics, L2-resident)
__global__ __launch_bounds__(256)
void edge_message_kernel(const float* __restrict__ edge_attr,
                         const __bf16* __restrict__ Wt,      // bf16 [H][32]
                         const float* __restrict__ edge_b,
                         const float* __restrict__ h,
                         const int* __restrict__ ei,         // [2,E]
                         float* __restrict__ agg, int E) {
    const int ED = 16, KP = 32;
    __shared__ __bf16 ldsA[64 * KP];                          // 4 KB
    __shared__ int s_src[64], s_dst[64];
    const int tid = threadIdx.x;
    const int eB  = blockIdx.x * 64;

    for (int idx = tid; idx < 64 * KP; idx += 256) {
        int r = idx >> 5;
        int k = idx & 31;
        float v = (k < ED) ? edge_attr[(size_t)(eB + r) * ED + k] : 0.0f;
        ldsA[idx] = (__bf16)v;
    }
    if (tid < 64) {
        s_src[tid] = ei[eB + tid];
        s_dst[tid] = ei[E + eB + tid];
    }
    __syncthreads();

    const int wave = tid >> 5, lane = tid & 31;
    const int half = lane >> 4, l16 = lane & 15;
    const int col  = wave * 16 + l16;

    bf16x16 b;
#pragma unroll
    for (int i = 0; i < 16; ++i)
        b[i] = Wt[(size_t)col * KP + 16 * half + i];
    const float bv = edge_b[col];

    for (int t0 = 0; t0 < 4; ++t0) {
        // Warm L2/L0 for the gather rows while the WMMA executes.
#pragma unroll
        for (int r = 0; r < 8; ++r) {
            int er = t0 * 16 + r + 8 * half;
            __builtin_prefetch(&h[(size_t)s_src[er] * HDIM + col], 0, 1);
        }
        bf16x16 a;
        f32x8 c = {};
#pragma unroll
        for (int i = 0; i < 16; ++i) {
            int ka = 16 * half + i;
            a[i] = ldsA[(t0 * 16 + l16) * KP + ka];
        }
        c = __builtin_amdgcn_wmma_f32_16x16x32_bf16(false, a, false, b, (short)0, c,
                                                    false, false);
#pragma unroll
        for (int r = 0; r < 8; ++r) {
            int er = t0 * 16 + r + 8 * half;
            float v = c[r] + bv + h[(size_t)s_src[er] * HDIM + col];
            v = fmaxf(v, 0.0f);
            atomicAdd(&agg[(size_t)s_dst[er] * HDIM + col], v);
        }
    }
}

// BatchNorm stats: per-column sum & sumsq over N rows (atomic partials).
__global__ __launch_bounds__(256)
void bn_stats_kernel(const float* __restrict__ z, float* __restrict__ stats, int N) {
    const int t = threadIdx.x;
    const int col = t & 127, rg = t >> 7;
    const int base = blockIdx.x * 256;
    float s = 0.0f, s2 = 0.0f;
    for (int r = base + rg; r < base + 256 && r < N; r += 2) {
        float v = z[(size_t)r * HDIM + col];
        s += v; s2 += v * v;
    }
    atomicAdd(&stats[col], s);
    atomicAdd(&stats[HDIM + col], s2);
}

__global__ __launch_bounds__(256)
void bn_apply_kernel(const float* __restrict__ z, const float* __restrict__ stats,
                     const float* __restrict__ gamma, const float* __restrict__ beta,
                     float* __restrict__ hout, int N) {
    size_t idx = (size_t)blockIdx.x * blockDim.x + threadIdx.x;
    if (idx >= (size_t)N * HDIM) return;
    int col = (int)(idx & 127);
    float invN = 1.0f / (float)N;
    float mu  = stats[col] * invN;
    float var = stats[HDIM + col] * invN - mu * mu;
    float v = (z[idx] - mu) * rsqrtf(var + 1e-5f) * gamma[col] + beta[col];
    hout[idx] = fmaxf(v, 0.0f);
}

__global__ __launch_bounds__(256)
void pool_accum_kernel(const float* __restrict__ h, const int* __restrict__ batch,
                       float* __restrict__ pool, float* __restrict__ cnt, int N) {
    size_t idx = (size_t)blockIdx.x * blockDim.x + threadIdx.x;
    if (idx >= (size_t)N * HDIM) return;
    int n = (int)(idx >> 7), col = (int)(idx & 127);
    int g = batch[n];
    atomicAdd(&pool[(size_t)g * HDIM + col], h[idx]);
    if (col == 0) atomicAdd(&cnt[g], 1.0f);
}

__global__ __launch_bounds__(256)
void pooled_div_kernel(const float* __restrict__ pool, const float* __restrict__ cnt,
                       float* __restrict__ pooled, int total) {
    int idx = blockIdx.x * blockDim.x + threadIdx.x;
    if (idx >= total) return;
    pooled[idx] = pool[idx] / fmaxf(cnt[idx >> 7], 1.0f);
}

extern "C" void kernel_launch(void* const* d_in, const int* in_sizes, int n_in,
                              void* d_out, int out_size, void* d_ws, size_t ws_size,
                              hipStream_t stream) {
    const int N = 100000, E = 1600000, L = 2, G = 256, H = HDIM;

    const float* x         = (const float*)d_in[0];
    const float* edge_attr = (const float*)d_in[1];
    const int*   edge_idx  = (const int*)d_in[2];
    const int*   batch_ids = (const int*)d_in[3];
    const float* node_w    = (const float*)d_in[4];
    const float* node_b    = (const float*)d_in[5];
    const float* edge_w    = (const float*)d_in[6];
    const float* edge_b    = (const float*)d_in[7];
    const float* mlp1_w    = (const float*)d_in[8];
    const float* mlp1_b    = (const float*)d_in[9];
    const float* mlp2_w    = (const float*)d_in[10];
    const float* mlp2_b    = (const float*)d_in[11];
    const float* bn_gamma  = (const float*)d_in[12];
    const float* bn_beta   = (const float*)d_in[13];
    const float* proj_w    = (const float*)d_in[14];
    const float* proj_b    = (const float*)d_in[15];
    float* out = (float*)d_out;

    char* ws = (char*)d_ws;
    size_t off = 0;
    auto alloc = [&](size_t bytes) -> void* {
        void* p = ws + off;
        off = (off + bytes + 255) & ~(size_t)255;
        return p;
    };
    float*  h      = (float*)alloc((size_t)N * H * 4);
    float*  agg    = (float*)alloc((size_t)N * H * 4);   // agg, then z
    float*  t      = (float*)alloc((size_t)N * H * 4);
    __bf16* nwt    = (__bf16*)alloc((size_t)H * 64 * 2);
    __bf16* ewt    = (__bf16*)alloc((size_t)H * 32 * 2);
    __bf16* m1t    = (__bf16*)alloc((size_t)L * H * H * 2);
    __bf16* m2t    = (__bf16*)alloc((size_t)L * H * H * 2);
    __bf16* pjt    = (__bf16*)alloc((size_t)H * H * 2);
    float*  stats  = (float*)alloc(2 * H * 4);
    float*  pool   = (float*)alloc((size_t)G * H * 4);
    float*  cnt    = (float*)alloc(G * 4);
    float*  pooled = (float*)alloc((size_t)G * H * 4);

    // Weight conversion (tiny).
    cvt_w_kernel<<<(H * 64 + 255) / 256, 256, 0, stream>>>(node_w, nwt, 64, 64);
    cvt_w_kernel<<<(H * 32 + 255) / 256, 256, 0, stream>>>(edge_w, ewt, 16, 32);
    for (int l = 0; l < L; ++l) {
        cvt_w_kernel<<<(H * H + 255) / 256, 256, 0, stream>>>(
            mlp1_w + (size_t)l * H * H, m1t + (size_t)l * H * H, H, H);
        cvt_w_kernel<<<(H * H + 255) / 256, 256, 0, stream>>>(
            mlp2_w + (size_t)l * H * H, m2t + (size_t)l * H * H, H, H);
    }
    cvt_w_kernel<<<(H * H + 255) / 256, 256, 0, stream>>>(proj_w, pjt, H, H);

    // h = x @ node_w + node_b
    gemm_wmma_kernel<<<(N + 63) / 64, 256, 0, stream>>>(x, nullptr, nwt, node_b,
                                                        h, N, 64, 0);

    const size_t NH4 = (size_t)N * H * 4;
    const int nhBlocks = (int)(((size_t)N * H + 255) / 256);
    for (int l = 0; l < L; ++l) {
        hipMemsetAsync(agg, 0, NH4, stream);
        edge_message_kernel<<<E / 64, 256, 0, stream>>>(edge_attr, ewt, edge_b, h,
                                                        edge_idx, agg, E);
        // t = relu((h+agg) @ mlp1 + b1)
        gemm_wmma_kernel<<<(N + 63) / 64, 256, 0, stream>>>(
            h, agg, m1t + (size_t)l * H * H, mlp1_b + l * H, t, N, H, 1);
        // z = t @ mlp2 + b2   (z overwrites agg)
        gemm_wmma_kernel<<<(N + 63) / 64, 256, 0, stream>>>(
            t, nullptr, m2t + (size_t)l * H * H, mlp2_b + l * H, agg, N, H, 0);
        hipMemsetAsync(stats, 0, 2 * H * 4, stream);
        bn_stats_kernel<<<(N + 255) / 256, 256, 0, stream>>>(agg, stats, N);
        bn_apply_kernel<<<nhBlocks, 256, 0, stream>>>(agg, stats, bn_gamma + l * H,
                                                      bn_beta + l * H, h, N);
    }

    hipMemsetAsync(pool, 0, (size_t)G * H * 4, stream);
    hipMemsetAsync(cnt, 0, G * 4, stream);
    pool_accum_kernel<<<nhBlocks, 256, 0, stream>>>(h, batch_ids, pool, cnt, N);
    pooled_div_kernel<<<(G * H + 255) / 256, 256, 0, stream>>>(pool, cnt, pooled, G * H);
    gemm_wmma_kernel<<<(G + 63) / 64, 256, 0, stream>>>(pooled, nullptr, pjt, proj_b,
                                                        out, G, H, 0);
}
